// GraphResBlock_91044716740921
// MI455X (gfx1250) — matrix operand order repaired and checked by reading, not compiled, not run on hardware
//
#include <hip/hip_runtime.h>
#include <hip/hip_bf16.h>

typedef __attribute__((ext_vector_type(16))) __bf16   v16bf;
typedef __attribute__((ext_vector_type(8)))  float    v8f;
typedef __attribute__((ext_vector_type(2)))  float    v2f;
typedef __attribute__((ext_vector_type(4)))  float    v4f;
typedef __attribute__((ext_vector_type(4)))  unsigned v4u;

#define NN      8192
#define IN_DIM  128
#define OUT_DIM 128
#define KCOLS   144       // 128 + 1 (ones col for deg) + 15 pad -> 9 WMMA col tiles
#define NTILES  9
#define SK      8         // split-K factor for occupancy

#define XT_BYTES ((size_t)KCOLS * NN * 2)          // 2.36 MB each (hi, lo)
#define P_BYTES  ((size_t)NN * KCOLS * 4)          // 4.72 MB
#define H_BYTES  ((size_t)NN * 256 * 4)            // 8.39 MB
#define WT_BYTES ((size_t)256 * OUT_DIM * 4)       // 128 KB

// ---------------------------------------------------------------------------
// Kernel 0a: build transposed, bf16 hi/lo split, padded B matrix
// XT[c][k]: c<128 -> X[k][c]; c==128 -> 1.0 (row-sum / deg column); else 0.
// ---------------------------------------------------------------------------
__global__ void prep_xt(const float* __restrict__ X,
                        unsigned short* __restrict__ xh,
                        unsigned short* __restrict__ xl) {
    int idx = blockIdx.x * 256 + threadIdx.x;       // KCOLS*NN / 256 blocks
    int c = idx >> 13;                              // / 8192
    int k = idx & (NN - 1);
    float v = 0.0f;
    if (c < IN_DIM)       v = X[(size_t)k * IN_DIM + c];
    else if (c == IN_DIM) v = 1.0f;
    unsigned u  = __float_as_uint(v);
    unsigned hi = u & 0xFFFF0000u;
    float    lo = v - __uint_as_float(hi);
    xh[idx] = (unsigned short)(u >> 16);
    xl[idx] = (unsigned short)(__float_as_uint(lo) >> 16);
}

// Kernel 0b: transpose W (256x128 -> WT[128][256]) so gemm2's B operand is one b64
__global__ void prep_wt(const float* __restrict__ W, float* __restrict__ WT) {
    int idx = blockIdx.x * 256 + threadIdx.x;       // 32768 threads
    int n = idx >> 8;
    int k = idx & 255;
    WT[idx] = W[(size_t)k * OUT_DIM + n];
}

// pack two f32 into a bf16 pair (truncate-hi) and the bf16 residual pair
__device__ inline void pack2(float f0, float f1, unsigned& h, unsigned& l) {
    unsigned u0 = __float_as_uint(f0), u1 = __float_as_uint(f1);
    h = __builtin_amdgcn_perm(u1, u0, 0x07060302u);       // {u1.hi16, u0.hi16}
    float l0 = f0 - __uint_as_float(u0 & 0xFFFF0000u);    // -> v_fma_mix_f32_bf16
    float l1 = f1 - __uint_as_float(u1 & 0xFFFF0000u);
    l = __builtin_amdgcn_perm(__float_as_uint(l1), __float_as_uint(l0), 0x07060302u);
}

// ---------------------------------------------------------------------------
// Kernel 1: P[8192][144] += A @ [X|1|0]  via split-bf16 WMMA (3 terms)
// grid = 64 row-groups * SK k-slices, block = 256 thr (8 waves),
// wave = one 16-row tile x all 9 col tiles, K-slice = 1024.
// ---------------------------------------------------------------------------
__global__ __launch_bounds__(256) void gemm1(const float* __restrict__ A,
                                             const unsigned short* __restrict__ xh,
                                             const unsigned short* __restrict__ xl,
                                             float* __restrict__ P) {
    const int wave = threadIdx.x >> 5;
    const int lane = threadIdx.x & 31;
    const int l16  = lane & 15;
    const int hl   = lane >> 4;                 // 0: lanes 0-15, 1: lanes 16-31
    const int g    = blockIdx.x & 63;           // row group
    const int ks   = blockIdx.x >> 6;           // k-slice 0..SK-1
    const int rowtile = g * 8 + wave;           // 0..511
    const float* arow = A + (size_t)(rowtile * 16 + l16) * NN;

    v8f acc[NTILES] = {};

    // K-invariant ones/pad tile (cols 128..143): col 128 = 1.0 (lo part exactly 0)
    union { unsigned u[8]; v16bf v; } B8;
    {
        const unsigned o = (l16 == 0) ? 0x3F803F80u : 0u;
#pragma unroll
        for (int i = 0; i < 8; ++i) B8.u[i] = o;
    }

    const int c0 = ks * (256 / SK);             // 32 chunks of K=32 per slice
    for (int kc = c0; kc < c0 + (256 / SK); ++kc) {
        const int kb = kc * 32;
        // ---- A operand: 16x32 f32 -> bf16 hi/lo, ISA A-layout (8-K lane swap)
        const int ka = kb + hl * 8;
        v4f q0 = __builtin_nontemporal_load(reinterpret_cast<const v4f*>(arow + ka));
        v4f q1 = __builtin_nontemporal_load(reinterpret_cast<const v4f*>(arow + ka + 4));
        v4f q2 = __builtin_nontemporal_load(reinterpret_cast<const v4f*>(arow + ka + 16));
        v4f q3 = __builtin_nontemporal_load(reinterpret_cast<const v4f*>(arow + ka + 20));
        union { unsigned u[8]; v16bf v; } Ah, Al;
        pack2(q0.x, q0.y, Ah.u[0], Al.u[0]);
        pack2(q0.z, q0.w, Ah.u[1], Al.u[1]);
        pack2(q1.x, q1.y, Ah.u[2], Al.u[2]);
        pack2(q1.z, q1.w, Ah.u[3], Al.u[3]);
        pack2(q2.x, q2.y, Ah.u[4], Al.u[4]);
        pack2(q2.z, q2.w, Ah.u[5], Al.u[5]);
        pack2(q3.x, q3.y, Ah.u[6], Al.u[6]);
        pack2(q3.z, q3.w, Ah.u[7], Al.u[7]);

        // ---- B operands straight from transposed bf16 arrays (L2/WGP$ hot)
        const int kx = kb + hl * 8;
#pragma unroll
        for (int t = 0; t < 8; ++t) {
            const size_t boff = (size_t)(t * 16 + l16) * NN + kx;
            union { v4u q[2]; v16bf v; } Bh, Bl;
            const v4u* bh = reinterpret_cast<const v4u*>(xh + boff);
            const v4u* bl = reinterpret_cast<const v4u*>(xl + boff);
            Bh.q[0] = bh[0]; Bh.q[1] = bh[2];
            Bl.q[0] = bl[0]; Bl.q[1] = bl[2];
            // error-compensated split: Ahi*Bhi + Ahi*Blo + Alo*Bhi
            acc[t] = __builtin_amdgcn_wmma_f32_16x16x32_bf16(false, Ah.v, false, Bh.v,
                                                             (short)0, acc[t], false, false);
            acc[t] = __builtin_amdgcn_wmma_f32_16x16x32_bf16(false, Ah.v, false, Bl.v,
                                                             (short)0, acc[t], false, false);
            acc[t] = __builtin_amdgcn_wmma_f32_16x16x32_bf16(false, Al.v, false, Bh.v,
                                                             (short)0, acc[t], false, false);
        }
        // ones/deg tile: B8 constant, lo(B8)==0 -> only two terms needed
        acc[8] = __builtin_amdgcn_wmma_f32_16x16x32_bf16(false, Ah.v, false, B8.v,
                                                         (short)0, acc[8], false, false);
        acc[8] = __builtin_amdgcn_wmma_f32_16x16x32_bf16(false, Al.v, false, B8.v,
                                                         (short)0, acc[8], false, false);
    }

    // ---- split-K reduction via f32 atomics (C/D layout: VGPR v -> M=v+8*hl)
    const int rbase = rowtile * 16 + hl * 8;
#pragma unroll
    for (int t = 0; t < NTILES; ++t) {
        const int col = t * 16 + l16;
#pragma unroll
        for (int v = 0; v < 8; ++v)
            unsafeAtomicAdd(&P[(size_t)(rbase + v) * KCOLS + col], acc[t][v]);
    }
}

// ---------------------------------------------------------------------------
// Kernel 2: H = [X | P[:, :128] / max(P[:,128], 1e-8)]
// ---------------------------------------------------------------------------
__global__ void build_h(const float* __restrict__ X,
                        const float* __restrict__ P,
                        float* __restrict__ H) {
    const int r = blockIdx.x;
    const int c = threadIdx.x;                  // 128 threads
    const float deg = fmaxf(P[(size_t)r * KCOLS + IN_DIM], 1e-8f);
    H[(size_t)r * 256 + c]           = X[(size_t)r * IN_DIM + c];
    H[(size_t)r * 256 + IN_DIM + c]  = P[(size_t)r * KCOLS + c] / deg;
}

// ---------------------------------------------------------------------------
// Kernel 3: out = X + gelu(H @ W + b), exact f32 via V_WMMA_F32_16X16X4_F32
// grid = 64, block = 256 (8 waves); wave = 16-row tile x 8 col tiles, K=256.
// ---------------------------------------------------------------------------
__global__ __launch_bounds__(256) void gemm2(const float* __restrict__ H,
                                             const float* __restrict__ WT,
                                             const float* __restrict__ b,
                                             const float* __restrict__ X,
                                             float* __restrict__ out) {
    const int wave = threadIdx.x >> 5;
    const int lane = threadIdx.x & 31;
    const int l16  = lane & 15;
    const int hl   = lane >> 4;
    const int rb   = (blockIdx.x * 8 + wave) * 16;

    v8f acc[8] = {};
    const float* hrow = H + (size_t)(rb + l16) * 256;

    for (int k = 0; k < 256; k += 4) {
        const int ka = k + hl * 2;              // f32 A layout: lanes 0-15 K0,K1; 16-31 K2,K3
        v2f a = *reinterpret_cast<const v2f*>(hrow + ka);
#pragma unroll
        for (int t = 0; t < 8; ++t) {
            // B from WT[n][k]: V0=W[ka][n], V1=W[ka+1][n] -> single aligned b64
            v2f bb = *reinterpret_cast<const v2f*>(WT + (size_t)(t * 16 + l16) * 256 + ka);
            acc[t] = __builtin_amdgcn_wmma_f32_16x16x4_f32(false, a, false, bb,
                                                           (short)0, acc[t], false, false);
        }
    }

#pragma unroll
    for (int t = 0; t < 8; ++t) {
        const int col  = t * 16 + l16;
        const float bi = b[col];
#pragma unroll
        for (int v = 0; v < 8; ++v) {
            const int row = rb + v + hl * 8;
            float y = acc[t][v] + bi;
            float g = 0.5f * y * (1.0f + tanhf(0.7978845608028654f * (y + 0.044715f * y * y * y)));
            out[(size_t)row * OUT_DIM + col] = X[(size_t)row * OUT_DIM + col] + g;
        }
    }
}

// ---------------------------------------------------------------------------
extern "C" void kernel_launch(void* const* d_in, const int* in_sizes, int n_in,
                              void* d_out, int out_size, void* d_ws, size_t ws_size,
                              hipStream_t stream) {
    const float* X = (const float*)d_in[0];
    const float* A = (const float*)d_in[1];
    const float* W = (const float*)d_in[2];
    const float* b = (const float*)d_in[3];
    float* out = (float*)d_out;

    char* ws = (char*)d_ws;
    unsigned short* xh = (unsigned short*)(ws);
    unsigned short* xl = (unsigned short*)(ws + XT_BYTES);
    float* P  = (float*)(ws + 2 * XT_BYTES);
    float* H  = (float*)(ws + 2 * XT_BYTES + P_BYTES);
    float* WT = (float*)(ws + 2 * XT_BYTES + P_BYTES + H_BYTES);

    hipMemsetAsync(P, 0, P_BYTES, stream);                        // split-K accumulator
    prep_xt<<<(KCOLS * NN) / 256, 256, 0, stream>>>(X, xh, xl);   // tiny one-time split
    prep_wt<<<(256 * OUT_DIM) / 256, 256, 0, stream>>>(W, WT);    // 128 KB W transpose
    gemm1<<<64 * SK, 256, 0, stream>>>(A, xh, xl, P);             // HBM-bound big GEMM
    build_h<<<NN, 128, 0, stream>>>(X, P, H);                     // normalize + concat
    gemm2<<<NN / 16 / 8, 256, 0, stream>>>(H, WT, b, X, out);     // small GEMM + gelu + res
}
